// AttentionBlock_33852932227102
// MI455X (gfx1250) — compile-verified, matrix operand order checked by reading, not run on hardware
//
#include <hip/hip_runtime.h>
#include <hip/hip_bf16.h>

// ---------------------------------------------------------------------------
// AttentionBlock for MI455X (gfx1250): GroupNorm + QKV + flash-attention +
// proj + residual.  All matmuls via v_wmma_f32_16x16x32_f16 (f32 accumulate).
// Flash-attention avoids materializing the 268MB [4,4096,4096] attn matrix,
// and stages K/V j-tiles into LDS once per workgroup with the CDNA5 async
// global->LDS path (ASYNCcnt), double-buffered.
// ---------------------------------------------------------------------------

#define BB 4
#define CCH 256
#define NN 4096            // 64*64
#define NPB (CCH * NN)     // elements per batch = 1048576

#define KSTR 264           // K LDS row stride in halves (528B: 16B-aligned, bank-spread)
#define VSTR 40            // V LDS row stride in halves (80B: 16B-aligned, bank-spread)

typedef __attribute__((ext_vector_type(16))) _Float16 v16h;
typedef __attribute__((ext_vector_type(8)))  _Float16 v8h;
typedef __attribute__((ext_vector_type(8)))  float    v8f;

#define WMMA16(a, b, c) \
  __builtin_amdgcn_wmma_f32_16x16x32_f16(false, (a), false, (b), (short)0, (c), false, false)

// Load one 16x32-half fragment (A layout, or B layout from a Bt row-major
// source).  Per CDNA5 ISA 7.12.2, lane (r,g) holds halves src[r][8g..8g+7]
// and src[r][16+8g..16+8g+7]  ->  two b128 loads per lane.
__device__ __forceinline__ v16h load_frag(const _Float16* __restrict__ base, int stride_h) {
  const int lane = threadIdx.x & 31;
  const int r = lane & 15, g = lane >> 4;
  const _Float16* p = base + (size_t)r * stride_h + 8 * g;
  union { v16h v; float4 q[2]; } u;
  u.q[0] = *(const float4*)(p);
  u.q[1] = *(const float4*)(p + 16);
  return u.v;
}

// CDNA5 async global->LDS 16B copy (VGLOBAL encoding, tracked by ASYNCcnt).
__device__ __forceinline__ void async_copy16(unsigned lds_off, const _Float16* g) {
  asm volatile("global_load_async_to_lds_b128 %0, %1, off"
               :: "v"(lds_off), "v"(g)
               : "memory");
}

// ------------------------- GroupNorm statistics ----------------------------
__global__ void __launch_bounds__(256) stats_partial_kernel(const float* __restrict__ x,
                                                            float* __restrict__ part) {
  __shared__ float ssum[256], ssq[256];
  const int b = blockIdx.x >> 8, j = blockIdx.x & 255;     // 256 blocks per batch
  const float* base = x + (size_t)b * NPB + (size_t)j * 4096;
  float s = 0.f, q = 0.f;
  for (int t = threadIdx.x; t < 4096; t += 256) { float v = base[t]; s += v; q += v * v; }
  ssum[threadIdx.x] = s; ssq[threadIdx.x] = q;
  __syncthreads();
  for (int st = 128; st > 0; st >>= 1) {
    if ((int)threadIdx.x < st) {
      ssum[threadIdx.x] += ssum[threadIdx.x + st];
      ssq[threadIdx.x]  += ssq[threadIdx.x + st];
    }
    __syncthreads();
  }
  if (threadIdx.x == 0) {
    part[blockIdx.x * 2 + 0] = ssum[0];
    part[blockIdx.x * 2 + 1] = ssq[0];
  }
}

__global__ void __launch_bounds__(256) stats_final_kernel(const float* __restrict__ part,
                                                          float* __restrict__ stats) {
  __shared__ float ssum[256], ssq[256];
  const int b = blockIdx.x;                                 // 4 blocks
  ssum[threadIdx.x] = part[(b * 256 + threadIdx.x) * 2 + 0];
  ssq[threadIdx.x]  = part[(b * 256 + threadIdx.x) * 2 + 1];
  __syncthreads();
  for (int st = 128; st > 0; st >>= 1) {
    if ((int)threadIdx.x < st) {
      ssum[threadIdx.x] += ssum[threadIdx.x + st];
      ssq[threadIdx.x]  += ssq[threadIdx.x + st];
    }
    __syncthreads();
  }
  if (threadIdx.x == 0) {
    const float inv = 1.f / (float)NPB;
    float mean = ssum[0] * inv;
    float var  = ssq[0] * inv - mean * mean;
    stats[b * 2 + 0] = mean;
    stats[b * 2 + 1] = rsqrtf(var + 1e-5f);
  }
}

// ------------- normalize + transpose x[b][c][n] -> Xnh[b][n][c] f16 --------
__global__ void __launch_bounds__(256) norm_transpose_kernel(const float* __restrict__ x,
                                                             const float* __restrict__ gamma,
                                                             const float* __restrict__ beta,
                                                             const float* __restrict__ stats,
                                                             _Float16* __restrict__ Xnh) {
  __shared__ float tile[32][33];
  int bid = blockIdx.x;
  const int nt = bid & 127; bid >>= 7;    // 128 n-tiles of 32
  const int ct = bid & 7;   bid >>= 3;    // 8 c-tiles of 32
  const int b  = bid;
  const int c0 = ct * 32, n0 = nt * 32;
  const float mean = stats[b * 2 + 0], rstd = stats[b * 2 + 1];
  const int col = threadIdx.x & 31, rowq = threadIdx.x >> 5;
#pragma unroll
  for (int p = 0; p < 4; ++p) {
    int row = rowq + p * 8;
    tile[row][col] = x[(size_t)(b * CCH + c0 + row) * NN + n0 + col];
  }
  __syncthreads();
#pragma unroll
  for (int p = 0; p < 4; ++p) {
    int nrow = rowq + p * 8;
    int c = c0 + col;
    float v = (tile[col][nrow] - mean) * rstd * gamma[c] + beta[c];
    Xnh[(size_t)(b * NN + n0 + nrow) * CCH + c] = (_Float16)v;
  }
}

__global__ void __launch_bounds__(256) convert_kernel(const float* __restrict__ src,
                                                      _Float16* __restrict__ dst, int n) {
  int i = blockIdx.x * 256 + threadIdx.x;
  if (i < n) dst[i] = (_Float16)src[i];
}

// ----------------------------- QKV GEMM (WMMA) -----------------------------
// D[d0:16][n0:16] = qkv_w[d,:] . xn[:,n] + bias; Q pre-scaled by C^-0.5.
__global__ void __launch_bounds__(256) qkv_gemm_kernel(const _Float16* __restrict__ Xnh,
                                                       const _Float16* __restrict__ Wh,
                                                       const float* __restrict__ bias,
                                                       _Float16* __restrict__ Qh,
                                                       _Float16* __restrict__ Kh,
                                                       _Float16* __restrict__ Vh) {
  const int wave = threadIdx.x >> 5, lane = threadIdx.x & 31;
  const int r = lane & 15, g = lane >> 4;
  int gw = blockIdx.x * 8 + wave;               // 4*48*256 = 49152 waves
  const int nt = gw & 255; gw >>= 8;
  const int dt = gw % 48;
  const int b  = gw / 48;
  const int n0 = nt * 16, d0 = dt * 16;

  v8f acc = {0.f, 0.f, 0.f, 0.f, 0.f, 0.f, 0.f, 0.f};
  const _Float16* ab = Wh  + (size_t)d0 * CCH;
  const _Float16* bb = Xnh + (size_t)(b * NN + n0) * CCH;
#pragma unroll
  for (int cc = 0; cc < 8; ++cc)
    acc = WMMA16(load_frag(ab + cc * 32, CCH), load_frag(bb + cc * 32, CCH), acc);

  float vals[8];
#pragma unroll
  for (int v = 0; v < 8; ++v) vals[v] = acc[v] + bias[d0 + v + 8 * g];

  if (d0 < 256) {                                // Q (fold in softmax scale)
    v8h hv;
#pragma unroll
    for (int v = 0; v < 8; ++v) hv[v] = (_Float16)(vals[v] * 0.0625f);
    *(v8h*)(Qh + (size_t)(b * NN + n0 + r) * CCH + d0 + 8 * g) = hv;
  } else if (d0 < 512) {                         // K
    v8h hv;
#pragma unroll
    for (int v = 0; v < 8; ++v) hv[v] = (_Float16)vals[v];
    *(v8h*)(Kh + (size_t)(b * NN + n0 + r) * CCH + (d0 - 256) + 8 * g) = hv;
  } else {                                       // V stored [b][c][n]
#pragma unroll
    for (int v = 0; v < 8; ++v)
      Vh[(size_t)(b * CCH + (d0 - 512) + v + 8 * g) * NN + n0 + r] = (_Float16)vals[v];
  }
}

// --------------------------- Flash attention -------------------------------
// One wave per 16-row i-tile; the workgroup (8 waves, 128 rows) cooperatively
// double-buffers each 32-wide K/V j-tile into LDS via async global->LDS
// loads (ASYNCcnt).  Per j-step: 16 WMMA for S = Q.K, online softmax with
// 16-lane shfl_xor row reductions, P transposed via a per-wave LDS slice
// (explicit s_wait_dscnt for the same-wave cross-lane RAW), then 16 WMMA
// accumulating O[16x256] in registers.
__global__ void __launch_bounds__(256) flash_attn_kernel(const _Float16* __restrict__ Qh,
                                                         const _Float16* __restrict__ Kh,
                                                         const _Float16* __restrict__ Vh,
                                                         _Float16* __restrict__ Oh) {
  __shared__ alignas(16) _Float16 kbuf[2][32 * KSTR];   // 2 x 16.5 KB
  __shared__ alignas(16) _Float16 vbuf[2][256 * VSTR];  // 2 x 20 KB
  __shared__ alignas(16) _Float16 pl[8][16 * 48];       // per-wave P slices

  const int tid = threadIdx.x;
  const int wave = tid >> 5, lane = tid & 31;
  const int r = lane & 15, g = lane >> 4;
  const int b  = blockIdx.x >> 5;                       // 32 blocks per batch
  const int i0 = (((blockIdx.x & 31) << 3) + wave) << 4;

  const _Float16* kgb = Kh + (size_t)b * NN * CCH;
  const _Float16* vgb = Vh + (size_t)b * CCH * NN;

  // ---- resident Q fragments (16 rows x 256 ch) ----
  v16h qf[8];
  const _Float16* qbase = Qh + (size_t)(b * NN + i0) * CCH;
#pragma unroll
  for (int cc = 0; cc < 8; ++cc) qf[cc] = load_frag(qbase + cc * 32, CCH);

  const v8f vzero = {0.f, 0.f, 0.f, 0.f, 0.f, 0.f, 0.f, 0.f};
  v8f oacc[16];
#pragma unroll
  for (int ch = 0; ch < 16; ++ch) oacc[ch] = vzero;
  float m[8], l[8];
#pragma unroll
  for (int v = 0; v < 8; ++v) { m[v] = -1e30f; l[v] = 0.f; }

  _Float16* myp = &pl[wave][0];

  // cooperative async stage of one 32-wide j-tile (8 x b128 per thread)
  auto stage = [&](int buf, int j0) {
    const unsigned kl = (unsigned)(uintptr_t)&kbuf[buf][0];
    const unsigned vl = (unsigned)(uintptr_t)&vbuf[buf][0];
#pragma unroll
    for (int u = 0; u < 4; ++u) {
      const int q = tid + u * 256;
      const int kr = q >> 5, kc = q & 31;     // K tile: 32 rows x 256 halves
      async_copy16(kl + (unsigned)(kr * KSTR + kc * 8) * 2,
                   kgb + (size_t)(j0 + kr) * CCH + kc * 8);
      const int vr = q >> 2, vc = q & 3;      // V tile: 256 rows x 32 halves
      async_copy16(vl + (unsigned)(vr * VSTR + vc * 8) * 2,
                   vgb + (size_t)vr * NN + j0 + vc * 8);
    }
  };

  stage(0, 0);
  for (int jt = 0; jt < NN / 32; ++jt) {
    const int cur = jt & 1;
    if (jt + 1 < NN / 32) {
      stage(cur ^ 1, (jt + 1) * 32);
      asm volatile("s_wait_asynccnt 0x8" ::: "memory");  // current tile landed
    } else {
      asm volatile("s_wait_asynccnt 0x0" ::: "memory");
    }
    __syncthreads();                                     // visible to all waves

    // ---- S tiles from LDS (scale already folded into Q) ----
    v8f s0 = vzero, s1 = vzero;
    const _Float16* kb0 = &kbuf[cur][0];
    const _Float16* kb1 = &kbuf[cur][16 * KSTR];
#pragma unroll
    for (int cc = 0; cc < 8; ++cc) {
      s0 = WMMA16(qf[cc], load_frag(kb0 + cc * 32, KSTR), s0);
      s1 = WMMA16(qf[cc], load_frag(kb1 + cc * 32, KSTR), s1);
    }
    // ---- online softmax; lane group g holds rows v+8g, col r ----
    float alpha[8], p0[8], p1[8];
#pragma unroll
    for (int v = 0; v < 8; ++v) {
      float mx = fmaxf(s0[v], s1[v]);
      mx = fmaxf(mx, __shfl_xor(mx, 1, 32));
      mx = fmaxf(mx, __shfl_xor(mx, 2, 32));
      mx = fmaxf(mx, __shfl_xor(mx, 4, 32));
      mx = fmaxf(mx, __shfl_xor(mx, 8, 32));
      float mnew = fmaxf(m[v], mx);
      alpha[v] = __expf(m[v] - mnew);
      p0[v] = __expf(s0[v] - mnew);
      p1[v] = __expf(s1[v] - mnew);
      float rs = p0[v] + p1[v];
      rs += __shfl_xor(rs, 1, 32);
      rs += __shfl_xor(rs, 2, 32);
      rs += __shfl_xor(rs, 4, 32);
      rs += __shfl_xor(rs, 8, 32);
      l[v] = l[v] * alpha[v] + rs;
      m[v] = mnew;
    }
#pragma unroll
    for (int ch = 0; ch < 16; ++ch)
#pragma unroll
      for (int v = 0; v < 8; ++v) oacc[ch][v] *= alpha[v];

    // ---- transpose P (C-layout -> A-layout) through per-wave LDS slice ----
#pragma unroll
    for (int v = 0; v < 8; ++v) {
      myp[(v + 8 * g) * 48 + r]      = (_Float16)p0[v];
      myp[(v + 8 * g) * 48 + 16 + r] = (_Float16)p1[v];
    }
    // same-wave cross-lane LDS RAW: DS is in-order per wave, but data must
    // land before re-load; the compiler cannot see the cross-lane dependency.
    asm volatile("s_wait_dscnt 0" ::: "memory");
    union { v16h v; float4 q[2]; } up;
    {
      const _Float16* pp = myp + r * 48 + 8 * g;
      up.q[0] = *(const float4*)(pp);
      up.q[1] = *(const float4*)(pp + 16);
    }
    // ---- O += P(16x32) x V^T(32xC) over 16 channel chunks (from LDS) ----
#pragma unroll
    for (int ch = 0; ch < 16; ++ch)
      oacc[ch] = WMMA16(up.v, load_frag(&vbuf[cur][(ch * 16) * VSTR], VSTR), oacc[ch]);

    __syncthreads();   // all waves done with buf[cur] before it is restaged
  }

  // ---- normalize and store O as [b][n][c] f16 ----
#pragma unroll
  for (int v = 0; v < 8; ++v) l[v] = 1.f / l[v];
  _Float16* ob = Oh + (size_t)(b * NN + i0) * CCH;
#pragma unroll
  for (int ch = 0; ch < 16; ++ch)
#pragma unroll
    for (int v = 0; v < 8; ++v)
      ob[(size_t)(v + 8 * g) * CCH + ch * 16 + r] = (_Float16)(oacc[ch][v] * l[v]);
}

// ---------------------- proj GEMM + bias + residual ------------------------
__global__ void __launch_bounds__(256) proj_kernel(const _Float16* __restrict__ Oh,
                                                   const _Float16* __restrict__ Wh,
                                                   const float* __restrict__ bias,
                                                   const float* __restrict__ x,
                                                   float* __restrict__ out) {
  const int wave = threadIdx.x >> 5, lane = threadIdx.x & 31;
  const int r = lane & 15, g = lane >> 4;
  int gw = blockIdx.x * 8 + wave;               // 4*16*256 = 16384 waves
  const int nt = gw & 255; gw >>= 8;
  const int dt = gw & 15;
  const int b  = gw >> 4;
  const int n0 = nt * 16, d0 = dt * 16;

  v8f acc = {0.f, 0.f, 0.f, 0.f, 0.f, 0.f, 0.f, 0.f};
  const _Float16* ab = Wh + (size_t)d0 * CCH;
  const _Float16* bb = Oh + (size_t)(b * NN + n0) * CCH;
#pragma unroll
  for (int cc = 0; cc < 8; ++cc)
    acc = WMMA16(load_frag(ab + cc * 32, CCH), load_frag(bb + cc * 32, CCH), acc);

#pragma unroll
  for (int v = 0; v < 8; ++v) {
    const int d = d0 + v + 8 * g;
    const size_t idx = (size_t)(b * CCH + d) * NN + n0 + r;
    out[idx] = acc[v] + bias[d] + x[idx];
  }
}

// ---------------------------------------------------------------------------
extern "C" void kernel_launch(void* const* d_in, const int* in_sizes, int n_in,
                              void* d_out, int out_size, void* d_ws, size_t ws_size,
                              hipStream_t stream) {
  const float* x      = (const float*)d_in[0];
  const float* gamma  = (const float*)d_in[1];
  const float* beta   = (const float*)d_in[2];
  const float* qkv_w  = (const float*)d_in[3];
  const float* qkv_b  = (const float*)d_in[4];
  const float* proj_w = (const float*)d_in[5];
  const float* proj_b = (const float*)d_in[6];
  float* out = (float*)d_out;

  // workspace layout (needs ~43 MB)
  char* ws = (char*)d_ws;
  float* stats = (float*)ws;                       // 8 floats
  float* part  = (float*)(ws + 256);               // 2048 floats
  const size_t HSZ = (size_t)BB * NN * CCH;        // f16 elements per tensor
  _Float16* Xnh    = (_Float16*)(ws + 65536);
  _Float16* Qh     = Xnh + HSZ;
  _Float16* Kh     = Qh + HSZ;
  _Float16* Vh     = Kh + HSZ;
  _Float16* Oh     = Vh + HSZ;
  _Float16* Wqkvh  = Oh + HSZ;                     // 768*256
  _Float16* Wprojh = Wqkvh + 768 * 256;            // 256*256

  stats_partial_kernel<<<BB * 256, 256, 0, stream>>>(x, part);
  stats_final_kernel<<<BB, 256, 0, stream>>>(part, stats);
  norm_transpose_kernel<<<BB * 8 * 128, 256, 0, stream>>>(x, gamma, beta, stats, Xnh);
  convert_kernel<<<(768 * 256 + 255) / 256, 256, 0, stream>>>(qkv_w, Wqkvh, 768 * 256);
  convert_kernel<<<(256 * 256 + 255) / 256, 256, 0, stream>>>(proj_w, Wprojh, 256 * 256);
  qkv_gemm_kernel<<<49152 / 8, 256, 0, stream>>>(Xnh, Wqkvh, qkv_b, Qh, Kh, Vh);
  flash_attn_kernel<<<1024 / 8, 256, 0, stream>>>(Qh, Kh, Vh, Oh);
  proj_kernel<<<16384 / 8, 256, 0, stream>>>(Oh, Wprojh, proj_b, x, out);
}
